// CalibyMPNN_2448131358767
// MI455X (gfx1250) — matrix operand order repaired and checked by reading, not compiled, not run on hardware
//
#include <hip/hip_runtime.h>

#define BSZ  2
#define NSEQ 2048
#define KNB  48
#define HD   128
#define CIN  384
#define FFD  512

typedef __attribute__((ext_vector_type(16))) __bf16        v16bf;
typedef __attribute__((ext_vector_type(8)))  float         v8f;
typedef __attribute__((ext_vector_type(4)))  unsigned int  v4u;
typedef __attribute__((ext_vector_type(2)))  unsigned int  v2u;
typedef __attribute__((ext_vector_type(4)))  float         v4f;

union Frag { v16bf v; v4u q[2]; };
union Pk4  { __bf16 e[4]; v2u u; };

__device__ __forceinline__ v8f zero8() {
  v8f z = {0.f,0.f,0.f,0.f,0.f,0.f,0.f,0.f};
  return z;
}

__device__ __forceinline__ v8f wmma_bf16(v16bf a, v16bf b, v8f c) {
  // D = A(16x32) * B(32x16) + C, fp32 accumulate
  return __builtin_amdgcn_wmma_f32_16x16x32_bf16(false, a, false, b, (short)0, c, false, false);
}

// A fragment (16x32 bf16) from LDS, row-major with leading dim lda.
// lane&15 = row, lane>>4 selects K-halves {h*8..h*8+7, 16+h*8..16+h*8+7}.
__device__ __forceinline__ v16bf load_a(const __bf16* A, int lda, int m0, int k0, int lane) {
  int row = m0 + (lane & 15);
  int h   = (lane >> 4) << 3;
  const __bf16* p = A + row * lda + k0 + h;
  Frag f;
  f.q[0] = *(const v4u*)(p);       // K = k0+h .. +7
  f.q[1] = *(const v4u*)(p + 16);  // K = k0+16+h .. +7
  return f.v;
}

// B fragment (32x16 bf16) from row-major bf16 weights [Kdim][ldn].
// lane&15 = N column, lane>>4 selects K-halves {h*8..h*8+7, 16+h*8..16+h*8+7}.
__device__ __forceinline__ v16bf load_b(const __bf16* W, int ldn, int k0, int n0, int lane) {
  const int h = (lane >> 4) << 3;
  const __bf16* p = W + (k0 + h) * ldn + n0 + (lane & 15);
  v16bf b;
  #pragma unroll
  for (int s = 0; s < 8; ++s) {
    b[s]     = p[s * ldn];
    b[8 + s] = p[(16 + s) * ldn];
  }
  return b;
}

__device__ __forceinline__ float gelu_f(float x) {
  return 0.5f * x * (1.0f + erff(x * 0.70710678118654752f));
}

__global__ void convert_bf16_kernel(const float* __restrict__ src,
                                    __bf16* __restrict__ dst, int n) {
  int i = blockIdx.x * 256 + threadIdx.x;
  if (i < n) dst[i] = (__bf16)src[i];
}

// ---------------- Kernel 1: node message + residual + LayerNorm ----------------
__global__ __launch_bounds__(256) void node_msg_kernel(
    const float* __restrict__ hV, const float* __restrict__ hE,
    const int*   __restrict__ Eidx, const float* __restrict__ mA,
    const __bf16* __restrict__ W1, const float* __restrict__ b1,
    const __bf16* __restrict__ W2, const float* __restrict__ b2,
    const __bf16* __restrict__ W3, const float* __restrict__ b3,
    const float* __restrict__ g1, const float* __restrict__ be1,
    float* __restrict__ hVmid)
{
  __shared__ __align__(16) char smem[62208];
  __bf16* Abuf = (__bf16*)smem;                 // [48][384] bf16
  __bf16* H1   = (__bf16*)(smem + 36864);       // [48][128] bf16
  __bf16* H2   = (__bf16*)(smem + 49152);       // [48][128] bf16
  float*  dh   = (float*) (smem + 61440);       // [128]
  float*  msk  = (float*) (smem + 61952);       // [48]
  float*  red  = (float*) (smem + 62144);       // [16]

  const int tid  = threadIdx.x;
  const int lane = tid & 31;
  const int wv   = tid >> 5;
  const int bn   = blockIdx.x;        // b*NSEQ + n
  const int bb   = bn >> 11;          // batch index (NSEQ=2048)

  if (tid < KNB) msk[tid] = mA[bn * KNB + tid];

  // Build concat A = [hV_expand | hE | gather(hV)] as bf16 in LDS
  for (int e4 = tid; e4 < KNB * (CIN / 4); e4 += 256) {
    int r = e4 / (CIN / 4);
    int c = (e4 - r * (CIN / 4)) * 4;
    const float* src;
    if (c < HD)            src = hV + bn * HD + c;
    else if (c < 2 * HD)   src = hE + (bn * KNB + r) * HD + (c - HD);
    else { int nb = Eidx[bn * KNB + r]; src = hV + (bb * NSEQ + nb) * HD + (c - 2 * HD); }
    v4f val = *(const v4f*)src;
    Pk4 p;
    p.e[0] = (__bf16)val[0]; p.e[1] = (__bf16)val[1];
    p.e[2] = (__bf16)val[2]; p.e[3] = (__bf16)val[3];
    *(v2u*)(Abuf + r * CIN + c) = p.u;
  }
  __syncthreads();

  const int n0  = wv << 4;          // this wave's 16 output columns
  const int col = lane & 15;
  const int ro  = (lane >> 4) << 3; // row offset 0 or 8 within M-tile

  // ---- GEMM1: [48,384] x [384,128] + GELU ----
  v8f a0 = zero8(), a1 = zero8(), a2 = zero8();
  #pragma unroll
  for (int k0 = 0; k0 < CIN; k0 += 32) {
    v16bf bf = load_b(W1, HD, k0, n0, lane);
    a0 = wmma_bf16(load_a(Abuf, CIN,  0, k0, lane), bf, a0);
    a1 = wmma_bf16(load_a(Abuf, CIN, 16, k0, lane), bf, a1);
    a2 = wmma_bf16(load_a(Abuf, CIN, 32, k0, lane), bf, a2);
  }
  {
    float bias = b1[n0 + col];
    #pragma unroll
    for (int r = 0; r < 8; ++r) {
      H1[(     ro + r) * HD + n0 + col] = (__bf16)gelu_f(a0[r] + bias);
      H1[(16 + ro + r) * HD + n0 + col] = (__bf16)gelu_f(a1[r] + bias);
      H1[(32 + ro + r) * HD + n0 + col] = (__bf16)gelu_f(a2[r] + bias);
    }
  }
  __syncthreads();

  // ---- GEMM2: [48,128] x [128,128] + GELU ----
  a0 = zero8(); a1 = zero8(); a2 = zero8();
  #pragma unroll
  for (int k0 = 0; k0 < HD; k0 += 32) {
    v16bf bf = load_b(W2, HD, k0, n0, lane);
    a0 = wmma_bf16(load_a(H1, HD,  0, k0, lane), bf, a0);
    a1 = wmma_bf16(load_a(H1, HD, 16, k0, lane), bf, a1);
    a2 = wmma_bf16(load_a(H1, HD, 32, k0, lane), bf, a2);
  }
  {
    float bias = b2[n0 + col];
    #pragma unroll
    for (int r = 0; r < 8; ++r) {
      H2[(     ro + r) * HD + n0 + col] = (__bf16)gelu_f(a0[r] + bias);
      H2[(16 + ro + r) * HD + n0 + col] = (__bf16)gelu_f(a1[r] + bias);
      H2[(32 + ro + r) * HD + n0 + col] = (__bf16)gelu_f(a2[r] + bias);
    }
  }
  __syncthreads();

  // ---- GEMM3: [48,128] x [128,128], masked sum over K=48 neighbors ----
  a0 = zero8(); a1 = zero8(); a2 = zero8();
  #pragma unroll
  for (int k0 = 0; k0 < HD; k0 += 32) {
    v16bf bf = load_b(W3, HD, k0, n0, lane);
    a0 = wmma_bf16(load_a(H2, HD,  0, k0, lane), bf, a0);
    a1 = wmma_bf16(load_a(H2, HD, 16, k0, lane), bf, a1);
    a2 = wmma_bf16(load_a(H2, HD, 32, k0, lane), bf, a2);
  }
  {
    float bias = b3[n0 + col];
    float part = 0.f;
    #pragma unroll
    for (int r = 0; r < 8; ++r) {
      part += (a0[r] + bias) * msk[     ro + r];
      part += (a1[r] + bias) * msk[16 + ro + r];
      part += (a2[r] + bias) * msk[32 + ro + r];
    }
    part += __shfl_down(part, 16, 32);   // combine column partials across half-waves
    if (lane < 16) dh[n0 + lane] = part;
  }
  __syncthreads();

  // ---- residual + LayerNorm over H=128 ----
  float x = 0.f;
  if (tid < HD) x = hV[bn * HD + tid] + dh[tid] * (1.0f / 30.0f);
  float s = x, s2 = x * x;
  #pragma unroll
  for (int o = 16; o > 0; o >>= 1) { s += __shfl_xor(s, o, 32); s2 += __shfl_xor(s2, o, 32); }
  if (lane == 0) { red[wv] = s; red[8 + wv] = s2; }
  __syncthreads();
  if (tid < HD) {
    float sum = red[0] + red[1] + red[2] + red[3];
    float sq  = red[8] + red[9] + red[10] + red[11];
    float m   = sum * (1.0f / 128.0f);
    float var = sq  * (1.0f / 128.0f) - m * m;
    hVmid[bn * HD + tid] = (x - m) * rsqrtf(var + 1e-5f) * g1[tid] + be1[tid];
  }
}

// ---------------- Kernel 2: position-wise FFN + residual + LN + mask ----------------
__global__ __launch_bounds__(256) void ffn_kernel(
    const float* __restrict__ hVmid, const float* __restrict__ mV,
    const __bf16* __restrict__ Wi, const float* __restrict__ bi,
    const __bf16* __restrict__ Wo, const float* __restrict__ bo,
    const float* __restrict__ g2, const float* __restrict__ be2,
    float* __restrict__ hVout)
{
  __shared__ __align__(16) char smem[28672];
  __bf16* XA   = (__bf16*)smem;                 // [16][128] bf16
  __bf16* Hh   = (__bf16*)(smem + 4096);        // [16][512] bf16
  float*  OutF = (float*) (smem + 20480);       // [16][128] fp32

  const int tid  = threadIdx.x;
  const int lane = tid & 31;
  const int wv   = tid >> 5;
  const int m0g  = blockIdx.x * 16;

  for (int e4 = tid; e4 < 16 * HD / 4; e4 += 256) {
    int idx = e4 * 4;
    v4f val = *(const v4f*)(hVmid + m0g * HD + idx);
    Pk4 p;
    p.e[0] = (__bf16)val[0]; p.e[1] = (__bf16)val[1];
    p.e[2] = (__bf16)val[2]; p.e[3] = (__bf16)val[3];
    *(v2u*)(XA + idx) = p.u;
  }
  __syncthreads();

  const int col = lane & 15;
  const int ro  = (lane >> 4) << 3;

  // ---- GEMM1: [16,128] x [128,512] + GELU. wave wv covers N-tiles wv+16j ----
  v8f acc[4] = { zero8(), zero8(), zero8(), zero8() };
  #pragma unroll
  for (int k0 = 0; k0 < HD; k0 += 32) {
    v16bf af = load_a(XA, HD, 0, k0, lane);
    #pragma unroll
    for (int j = 0; j < 4; ++j) {
      int n0 = (wv << 4) + j * 256;
      acc[j] = wmma_bf16(af, load_b(Wi, FFD, k0, n0, lane), acc[j]);
    }
  }
  #pragma unroll
  for (int j = 0; j < 4; ++j) {
    int n0 = (wv << 4) + j * 256;
    float bias = bi[n0 + col];
    #pragma unroll
    for (int r = 0; r < 8; ++r)
      Hh[(ro + r) * FFD + n0 + col] = (__bf16)gelu_f(acc[j][r] + bias);
  }
  __syncthreads();

  // ---- GEMM2: [16,512] x [512,128] + bias + residual ----
  v8f a = zero8();
  const int n0 = wv << 4;
  #pragma unroll
  for (int k0 = 0; k0 < FFD; k0 += 32)
    a = wmma_bf16(load_a(Hh, FFD, 0, k0, lane), load_b(Wo, HD, k0, n0, lane), a);
  {
    float bias = bo[n0 + col];
    #pragma unroll
    for (int r = 0; r < 8; ++r)
      OutF[(ro + r) * HD + n0 + col] =
          a[r] + bias + hVmid[(m0g + ro + r) * HD + n0 + col];
  }
  __syncthreads();

  // ---- per-row LN + mask_V: wave handles 2 rows ----
  #pragma unroll
  for (int rr = 0; rr < 2; ++rr) {
    int row = wv * 2 + rr;
    float x0 = OutF[row * HD + lane];
    float x1 = OutF[row * HD + lane + 32];
    float x2 = OutF[row * HD + lane + 64];
    float x3 = OutF[row * HD + lane + 96];
    float s  = x0 + x1 + x2 + x3;
    float s2 = x0*x0 + x1*x1 + x2*x2 + x3*x3;
    #pragma unroll
    for (int o = 16; o > 0; o >>= 1) { s += __shfl_xor(s, o, 32); s2 += __shfl_xor(s2, o, 32); }
    float m   = s  * (1.0f / 128.0f);
    float var = s2 * (1.0f / 128.0f) - m * m;
    float rs  = rsqrtf(var + 1e-5f);
    float mk  = mV[m0g + row];
    float* op = hVout + (m0g + row) * HD;
    op[lane     ] = mk * ((x0 - m) * rs * g2[lane     ] + be2[lane     ]);
    op[lane + 32] = mk * ((x1 - m) * rs * g2[lane + 32] + be2[lane + 32]);
    op[lane + 64] = mk * ((x2 - m) * rs * g2[lane + 64] + be2[lane + 64]);
    op[lane + 96] = mk * ((x3 - m) * rs * g2[lane + 96] + be2[lane + 96]);
  }
}

// ---------------- Kernel 3: edge update + residual + per-edge LN ----------------
__global__ __launch_bounds__(256) void edge_kernel(
    const float* __restrict__ hVf, const float* __restrict__ hE,
    const int*   __restrict__ Eidx,
    const __bf16* __restrict__ W11, const float* __restrict__ b11,
    const __bf16* __restrict__ W12, const float* __restrict__ b12,
    const __bf16* __restrict__ W13, const float* __restrict__ b13,
    const float* __restrict__ g3, const float* __restrict__ be3,
    float* __restrict__ hEout)
{
  __shared__ __align__(16) char smem[61440];
  __bf16* Abuf = (__bf16*)smem;                 // [48][384] bf16
  float*  OutF = (float*)smem;                  // alias: [48][128] fp32 (24576B)
  __bf16* H1   = (__bf16*)(smem + 36864);
  __bf16* H2   = (__bf16*)(smem + 49152);

  const int tid  = threadIdx.x;
  const int lane = tid & 31;
  const int wv   = tid >> 5;
  const int bn   = blockIdx.x;
  const int bb   = bn >> 11;

  for (int e4 = tid; e4 < KNB * (CIN / 4); e4 += 256) {
    int r = e4 / (CIN / 4);
    int c = (e4 - r * (CIN / 4)) * 4;
    const float* src;
    if (c < HD)          src = hVf + bn * HD + c;
    else if (c < 2 * HD) src = hE + (bn * KNB + r) * HD + (c - HD);
    else { int nb = Eidx[bn * KNB + r]; src = hVf + (bb * NSEQ + nb) * HD + (c - 2 * HD); }
    v4f val = *(const v4f*)src;
    Pk4 p;
    p.e[0] = (__bf16)val[0]; p.e[1] = (__bf16)val[1];
    p.e[2] = (__bf16)val[2]; p.e[3] = (__bf16)val[3];
    *(v2u*)(Abuf + r * CIN + c) = p.u;
  }
  __syncthreads();

  const int n0  = wv << 4;
  const int col = lane & 15;
  const int ro  = (lane >> 4) << 3;

  v8f a0 = zero8(), a1 = zero8(), a2 = zero8();
  #pragma unroll
  for (int k0 = 0; k0 < CIN; k0 += 32) {
    v16bf bf = load_b(W11, HD, k0, n0, lane);
    a0 = wmma_bf16(load_a(Abuf, CIN,  0, k0, lane), bf, a0);
    a1 = wmma_bf16(load_a(Abuf, CIN, 16, k0, lane), bf, a1);
    a2 = wmma_bf16(load_a(Abuf, CIN, 32, k0, lane), bf, a2);
  }
  {
    float bias = b11[n0 + col];
    #pragma unroll
    for (int r = 0; r < 8; ++r) {
      H1[(     ro + r) * HD + n0 + col] = (__bf16)gelu_f(a0[r] + bias);
      H1[(16 + ro + r) * HD + n0 + col] = (__bf16)gelu_f(a1[r] + bias);
      H1[(32 + ro + r) * HD + n0 + col] = (__bf16)gelu_f(a2[r] + bias);
    }
  }
  __syncthreads();

  a0 = zero8(); a1 = zero8(); a2 = zero8();
  #pragma unroll
  for (int k0 = 0; k0 < HD; k0 += 32) {
    v16bf bf = load_b(W12, HD, k0, n0, lane);
    a0 = wmma_bf16(load_a(H1, HD,  0, k0, lane), bf, a0);
    a1 = wmma_bf16(load_a(H1, HD, 16, k0, lane), bf, a1);
    a2 = wmma_bf16(load_a(H1, HD, 32, k0, lane), bf, a2);
  }
  {
    float bias = b12[n0 + col];
    #pragma unroll
    for (int r = 0; r < 8; ++r) {
      H2[(     ro + r) * HD + n0 + col] = (__bf16)gelu_f(a0[r] + bias);
      H2[(16 + ro + r) * HD + n0 + col] = (__bf16)gelu_f(a1[r] + bias);
      H2[(32 + ro + r) * HD + n0 + col] = (__bf16)gelu_f(a2[r] + bias);
    }
  }
  __syncthreads();

  a0 = zero8(); a1 = zero8(); a2 = zero8();
  #pragma unroll
  for (int k0 = 0; k0 < HD; k0 += 32) {
    v16bf bf = load_b(W13, HD, k0, n0, lane);
    a0 = wmma_bf16(load_a(H2, HD,  0, k0, lane), bf, a0);
    a1 = wmma_bf16(load_a(H2, HD, 16, k0, lane), bf, a1);
    a2 = wmma_bf16(load_a(H2, HD, 32, k0, lane), bf, a2);
  }
  {
    float bias = b13[n0 + col];
    #pragma unroll
    for (int r = 0; r < 8; ++r) {
      int r0 = ro + r, r1 = 16 + ro + r, r2 = 32 + ro + r;
      OutF[r0 * HD + n0 + col] = a0[r] + bias + hE[(bn * KNB + r0) * HD + n0 + col];
      OutF[r1 * HD + n0 + col] = a1[r] + bias + hE[(bn * KNB + r1) * HD + n0 + col];
      OutF[r2 * HD + n0 + col] = a2[r] + bias + hE[(bn * KNB + r2) * HD + n0 + col];
    }
  }
  __syncthreads();

  // per-edge-row LN: wave handles rows wv*6 .. wv*6+5
  #pragma unroll
  for (int rr = 0; rr < 6; ++rr) {
    int row = wv * 6 + rr;
    float x0 = OutF[row * HD + lane];
    float x1 = OutF[row * HD + lane + 32];
    float x2 = OutF[row * HD + lane + 64];
    float x3 = OutF[row * HD + lane + 96];
    float s  = x0 + x1 + x2 + x3;
    float s2 = x0*x0 + x1*x1 + x2*x2 + x3*x3;
    #pragma unroll
    for (int o = 16; o > 0; o >>= 1) { s += __shfl_xor(s, o, 32); s2 += __shfl_xor(s2, o, 32); }
    float m   = s  * (1.0f / 128.0f);
    float var = s2 * (1.0f / 128.0f) - m * m;
    float rs  = rsqrtf(var + 1e-5f);
    float* op = hEout + (bn * KNB + row) * HD;
    op[lane     ] = (x0 - m) * rs * g3[lane     ] + be3[lane     ];
    op[lane + 32] = (x1 - m) * rs * g3[lane + 32] + be3[lane + 32];
    op[lane + 64] = (x2 - m) * rs * g3[lane + 64] + be3[lane + 64];
    op[lane + 96] = (x3 - m) * rs * g3[lane + 96] + be3[lane + 96];
  }
}

extern "C" void kernel_launch(void* const* d_in, const int* in_sizes, int n_in,
                              void* d_out, int out_size, void* d_ws, size_t ws_size,
                              hipStream_t stream) {
  (void)in_sizes; (void)n_in; (void)out_size; (void)ws_size;
  const float* hV  = (const float*)d_in[0];
  const float* hE  = (const float*)d_in[1];
  const int*   Eix = (const int*)  d_in[2];
  const float* mV  = (const float*)d_in[3];
  const float* mA  = (const float*)d_in[4];
  const float* W1  = (const float*)d_in[5];  const float* b1  = (const float*)d_in[6];
  const float* W2  = (const float*)d_in[7];  const float* b2  = (const float*)d_in[8];
  const float* W3  = (const float*)d_in[9];  const float* b3  = (const float*)d_in[10];
  const float* Wdi = (const float*)d_in[11]; const float* bdi = (const float*)d_in[12];
  const float* Wdo = (const float*)d_in[13]; const float* bdo = (const float*)d_in[14];
  const float* W11 = (const float*)d_in[15]; const float* b11 = (const float*)d_in[16];
  const float* W12 = (const float*)d_in[17]; const float* b12 = (const float*)d_in[18];
  const float* W13 = (const float*)d_in[19]; const float* b13 = (const float*)d_in[20];
  const float* g1  = (const float*)d_in[21]; const float* be1 = (const float*)d_in[22];
  const float* g2  = (const float*)d_in[23]; const float* be2 = (const float*)d_in[24];
  const float* g3  = (const float*)d_in[25]; const float* be3 = (const float*)d_in[26];

  // workspace layout: bf16 weight copies, then h_V_mid fp32
  __bf16* wb   = (__bf16*)d_ws;
  __bf16* wW1  = wb;            // 384*128
  __bf16* wW2  = wb + 49152;    // 128*128
  __bf16* wW3  = wb + 65536;    // 128*128
  __bf16* wWi  = wb + 81920;    // 128*512
  __bf16* wWo  = wb + 147456;   // 512*128
  __bf16* wW11 = wb + 212992;   // 384*128
  __bf16* wW12 = wb + 262144;   // 128*128
  __bf16* wW13 = wb + 278528;   // 128*128 (ends at 294912 elems = 589824 B)
  float*  hVmid = (float*)((char*)d_ws + 589824);   // 2*2048*128 fp32

  float* hVout = (float*)d_out;
  float* hEout = hVout + BSZ * NSEQ * HD;

  struct CV { const float* s; __bf16* d; int n; } cvs[8] = {
    {W1,  wW1,  CIN*HD}, {W2,  wW2,  HD*HD}, {W3,  wW3,  HD*HD},
    {Wdi, wWi,  HD*FFD}, {Wdo, wWo,  FFD*HD},
    {W11, wW11, CIN*HD}, {W12, wW12, HD*HD}, {W13, wW13, HD*HD}
  };
  for (int i = 0; i < 8; ++i)
    convert_bf16_kernel<<<(cvs[i].n + 255) / 256, 256, 0, stream>>>(cvs[i].s, cvs[i].d, cvs[i].n);

  node_msg_kernel<<<BSZ * NSEQ, 256, 0, stream>>>(
      hV, hE, Eix, mA, wW1, b1, wW2, b2, wW3, b3, g1, be1, hVmid);

  ffn_kernel<<<(BSZ * NSEQ) / 16, 256, 0, stream>>>(
      hVmid, mV, wWi, bdi, wWo, bdo, g2, be2, hVout);

  edge_kernel<<<BSZ * NSEQ, 256, 0, stream>>>(
      hVout, hE, Eix, wW11, b11, wW12, b12, wW13, b13, g3, be3, hEout);
}
